// MAHADecoderBlock_20804821581903
// MI455X (gfx1250) — compile-verified, matrix operand order checked by reading, not run on hardware
//
#include <hip/hip_runtime.h>
#include <stdint.h>
#include <stddef.h>

typedef __bf16 bf16x16 __attribute__((ext_vector_type(16)));
typedef float  f32x8   __attribute__((ext_vector_type(8)));

#define DEVINL __device__ __forceinline__

constexpr int    cB    = 2;
constexpr int    cN    = 2048;
constexpr int    cD    = 1024;
constexpr int    cH    = 16;
constexpr int    cDH   = 64;
constexpr int    cHID  = 2734;
constexpr int    cHIDP = 2752;           // padded to multiple of 32
constexpr int    cBN   = cB * cN;        // 4096
constexpr size_t cBND  = (size_t)cB * cN * cD;  // 4194304

// ---------------- workspace layout (bytes) ----------------
constexpr size_t SZ_WDEC = (size_t)3 * cD * cD * 2;
constexpr size_t SZ_WDD  = (size_t)cD * cD * 2;
constexpr size_t SZ_WGU  = (size_t)cD * cHID * 2;
constexpr size_t SZ_WDN  = (size_t)cHIDP * cD * 2;

constexpr size_t OFF_WDEC = 0;
constexpr size_t OFF_WQ   = OFF_WDEC + SZ_WDEC;
constexpr size_t OFF_WK   = OFF_WQ + SZ_WDEC;
constexpr size_t OFF_WV   = OFF_WK + SZ_WDEC;
constexpr size_t OFF_WO   = OFF_WV + SZ_WDD;
constexpr size_t OFF_WG   = OFF_WO + SZ_WDD;
constexpr size_t OFF_WU   = OFF_WG + SZ_WGU;
constexpr size_t OFF_WD   = OFF_WU + SZ_WGU;
constexpr size_t OFF_XN0  = OFF_WD + SZ_WDN;
constexpr size_t OFF_S1   = OFF_XN0 + cBND * 2;
constexpr size_t OFF_S2   = OFF_S1 + (size_t)cB * 1024 * cD * 2;
constexpr size_t OFF_X1   = OFF_S2 + (size_t)cB * 512 * cD * 2;
constexpr size_t OFF_XN2  = OFF_X1 + cBND * 4;
constexpr size_t OFF_G    = OFF_XN2 + cBND * 2;
constexpr size_t OFF_SC   = OFF_G + 256;
constexpr size_t OFF_UN   = OFF_SC + 256;
// union region A: attention-phase temporaries
constexpr size_t TOKS     = 7168;        // 4096 + 2048 + 1024 token-rows
constexpr size_t SZ_TOKBF = TOKS * cD * 2;
constexpr size_t OFF_XDEC = OFF_UN;
constexpr size_t OFF_Q    = OFF_XDEC + SZ_TOKBF;
constexpr size_t OFF_K    = OFF_Q + SZ_TOKBF;
constexpr size_t OFF_V    = OFF_K + SZ_TOKBF;
constexpr size_t OFF_O    = OFF_V + SZ_TOKBF;
constexpr size_t OFF_Z    = OFF_O + SZ_TOKBF;           // f32, TOKS*cD*4
// union region B: FFN-phase temporaries (reuses region A after mix1)
constexpr size_t OFF_GATE = OFF_UN;
constexpr size_t OFF_UP   = OFF_GATE + (size_t)cBN * cHID * 4;
constexpr size_t OFF_HB   = OFF_UP + (size_t)cBN * cHID * 4;
constexpr size_t OFF_FFN  = OFF_HB + (size_t)cBN * cHIDP * 2;

// ---------------- helpers ----------------
DEVINL uint16_t f2bf(float f) {
  uint32_t u = __builtin_bit_cast(uint32_t, f);
  u += 0x7FFFu + ((u >> 16) & 1u);     // round-to-nearest-even
  return (uint16_t)(u >> 16);
}
DEVINL float bf2f(uint16_t h) {
  uint32_t u = ((uint32_t)h) << 16;
  return __builtin_bit_cast(float, u);
}

union Frag {
  bf16x16  v;
  uint16_t h[16];
  uint32_t w[8];
};

DEVINL f32x8 zero8() {
  f32x8 z;
#pragma unroll
  for (int i = 0; i < 8; ++i) z[i] = 0.0f;
  return z;
}

// cross-lane xor reductions within 16-lane groups (ds_swizzle: one DS op, no addr VALU)
template <int XM>
DEVINL float swz(float x) {
#if __has_builtin(__builtin_amdgcn_ds_swizzle)
  return __builtin_bit_cast(
      float, __builtin_amdgcn_ds_swizzle(__builtin_bit_cast(int, x), (XM << 10) | 0x1f));
#else
  return __shfl_xor(x, XM, 32);
#endif
}
DEVINL float redmax16(float x) {
  x = fmaxf(x, swz<1>(x)); x = fmaxf(x, swz<2>(x));
  x = fmaxf(x, swz<4>(x)); x = fmaxf(x, swz<8>(x));
  return x;
}
DEVINL float redsum16(float x) {
  x += swz<1>(x); x += swz<2>(x); x += swz<4>(x); x += swz<8>(x);
  return x;
}

// ---------------- conversion kernels ----------------
__global__ __launch_bounds__(256) void k_conv_bf16(const float* __restrict__ src,
                                                   uint16_t* __restrict__ dst, size_t n) {
  size_t stride = (size_t)gridDim.x * blockDim.x;
  for (size_t i = (size_t)blockIdx.x * blockDim.x + threadIdx.x; i < n; i += stride)
    dst[i] = f2bf(src[i]);
}

__global__ __launch_bounds__(256) void k_conv_pad_rows(const float* __restrict__ src,
                                                       uint16_t* __restrict__ dst,
                                                       int rows, int cols, int rowsPad) {
  size_t n = (size_t)rowsPad * cols;
  size_t stride = (size_t)gridDim.x * blockDim.x;
  for (size_t i = (size_t)blockIdx.x * blockDim.x + threadIdx.x; i < n; i += stride) {
    int r = (int)(i / cols);
    dst[i] = (r < rows) ? f2bf(src[i]) : (uint16_t)0;
  }
}

// ---------------- RMSNorm (one block per row of D=1024) ----------------
__global__ __launch_bounds__(256) void k_rmsnorm_bf16(const float* __restrict__ x,
                                                      const float* __restrict__ w,
                                                      uint16_t* __restrict__ out) {
  __shared__ float red[256];
  size_t row = blockIdx.x;
  const float* xr = x + row * cD;
  float ss = 0.0f;
  for (int j = threadIdx.x; j < cD; j += 256) { float v = xr[j]; ss += v * v; }
  red[threadIdx.x] = ss;
  __syncthreads();
  for (int s = 128; s > 0; s >>= 1) {
    if ((int)threadIdx.x < s) red[threadIdx.x] += red[threadIdx.x + s];
    __syncthreads();
  }
  float inv = rsqrtf(red[0] / (float)cD + 1e-6f);
  for (int j = threadIdx.x; j < cD; j += 256)
    out[row * cD + j] = f2bf(xr[j] * w[j] * inv);
}

// ---------------- avg-pool by 2 along sequence (bf16) ----------------
__global__ __launch_bounds__(256) void k_pool2(const uint16_t* __restrict__ in,
                                               uint16_t* __restrict__ out,
                                               int nOut, size_t total) {
  size_t stride = (size_t)gridDim.x * blockDim.x;
  for (size_t i = (size_t)blockIdx.x * blockDim.x + threadIdx.x; i < total; i += stride) {
    size_t b = i / ((size_t)nOut * cD);
    size_t rem = i - b * (size_t)nOut * cD;
    size_t r = rem / cD, d = rem - r * cD;
    size_t base = (b * (size_t)(2 * nOut) + 2 * r) * cD + d;
    out[i] = f2bf(0.5f * (bf2f(in[base]) + bf2f(in[base + cD])));
  }
}

// ---------------- tiled WMMA GEMM: C[M,Nc] = A[M,K](bf16) * B[K,Nc](bf16) ----------------
// block: 256 threads = 8 waves; block tile 128x128; wave tile 32x64 (2x4 WMMA tiles)
// LDS: A tile 128x32 bf16; B tile packed as k-pair dwords [16][128] so every
// B-fragment register is a single aligned ds_load_b32.
template <bool OUT_BF16>
__global__ __launch_bounds__(256) void k_gemm_wmma(const uint16_t* __restrict__ A,
                                                   const uint16_t* __restrict__ Bw,
                                                   void* __restrict__ Cout,
                                                   int M, int K, int Nc) {
  __shared__ uint16_t lA[128 * 32];
  __shared__ uint32_t lBp[16 * 128];
  const int tid  = threadIdx.x;
  const int lane = tid & 31;
  const int wave = tid >> 5;
  const int wm = wave & 3, wn = wave >> 2;   // 4x2 wave grid
  const int rowBase = blockIdx.x * 128;
  const int colBase = blockIdx.y * 128;
  const int ln = lane & 15, lh = lane >> 4;
  const bool afull = (rowBase + 128 <= M);
  const bool bfull = (colBase + 128 <= Nc);

  // per-thread load coordinates
  const int ar = tid >> 1, akoff = (tid & 1) * 16;   // A: 16 bf16 per thread
  const int br2 = tid >> 4, bcb = (tid & 15) * 8;    // B: 2 rows x 8 cols per thread

  f32x8 acc[2][4];
#pragma unroll
  for (int r = 0; r < 2; ++r)
#pragma unroll
    for (int c = 0; c < 4; ++c) acc[r][c] = zero8();

  for (int k0 = 0; k0 < K; k0 += 32) {
    // ---- A tile 128x32 ----
    {
      int gr = rowBase + ar;
      uint4 v0 = make_uint4(0, 0, 0, 0), v1 = make_uint4(0, 0, 0, 0);
      if (afull || gr < M) {
        const uint4* src = (const uint4*)(A + (size_t)gr * K + k0 + akoff);
        v0 = src[0]; v1 = src[1];
      }
      uint4* pl = (uint4*)&lA[ar * 32 + akoff];
      pl[0] = v0; pl[1] = v1;
    }
    // ---- B tile 32x128, packed into k-pair dwords ----
    {
      const uint16_t* p0 = Bw + (size_t)(k0 + 2 * br2) * Nc + colBase + bcb;
      const uint16_t* p1 = p0 + Nc;
      uint32_t* dst = &lBp[br2 * 128 + bcb];
      if (bfull) {
#pragma unroll
        for (int t = 0; t < 4; ++t) {
          uint32_t a01 = *(const uint32_t*)(p0 + 2 * t);
          uint32_t b01 = *(const uint32_t*)(p1 + 2 * t);
          dst[2 * t]     = (a01 & 0xffffu) | (b01 << 16);
          dst[2 * t + 1] = (a01 >> 16) | (b01 & 0xffff0000u);
        }
      } else {
#pragma unroll
        for (int j = 0; j < 8; ++j) {
          int gc = colBase + bcb + j;
          uint32_t lo = 0, hi = 0;
          if (gc < Nc) { lo = p0[j]; hi = p1[j]; }
          dst[j] = lo | (hi << 16);
        }
      }
    }
    // prefetch next k-tile
    if (k0 + 32 < K) {
      if (afull) __builtin_prefetch(A + (size_t)(rowBase + ar) * K + (k0 + 32) + akoff, 0, 0);
      if (bfull) __builtin_prefetch(Bw + (size_t)(k0 + 32 + 2 * br2) * Nc + colBase + bcb, 0, 0);
    }
    __syncthreads();

    Frag af[2], bfr[4];
#pragma unroll
    for (int r = 0; r < 2; ++r) {
      int m = 32 * wm + 16 * r + ln;
      const uint4* pa = (const uint4*)&lA[m * 32 + lh * 8];
      uint4 a0 = pa[0];                          // kk = lh*8 + {0,2,4,6}
      uint4 a1 = *(const uint4*)&lA[m * 32 + 16 + lh * 8];   // kk = 16 + ...
      af[r].w[0] = a0.x; af[r].w[1] = a0.y; af[r].w[2] = a0.z; af[r].w[3] = a0.w;
      af[r].w[4] = a1.x; af[r].w[5] = a1.y; af[r].w[6] = a1.z; af[r].w[7] = a1.w;
    }
#pragma unroll
    for (int c = 0; c < 4; ++c) {
      int nn = 64 * wn + 16 * c + ln;
#pragma unroll
      for (int j = 0; j < 8; ++j) {
        int kp = (j >> 2) * 8 + lh * 4 + (j & 3);  // pair index = kk/2
        bfr[c].w[j] = lBp[kp * 128 + nn];
      }
    }
#pragma unroll
    for (int r = 0; r < 2; ++r)
#pragma unroll
      for (int c = 0; c < 4; ++c)
        acc[r][c] = __builtin_amdgcn_wmma_f32_16x16x32_bf16(
            false, af[r].v, false, bfr[c].v, (short)0, acc[r][c], false, false);
    __syncthreads();
  }

  if (afull && bfull) {
#pragma unroll
    for (int r = 0; r < 2; ++r)
#pragma unroll
      for (int c = 0; c < 4; ++c)
#pragma unroll
        for (int i = 0; i < 8; ++i) {
          int m  = rowBase + 32 * wm + 16 * r + i + 8 * lh;
          int nc = colBase + 64 * wn + 16 * c + ln;
          if (OUT_BF16)
            ((uint16_t*)Cout)[(size_t)m * Nc + nc] = f2bf(acc[r][c][i]);
          else
            ((float*)Cout)[(size_t)m * Nc + nc] = acc[r][c][i];
        }
  } else {
#pragma unroll
    for (int r = 0; r < 2; ++r)
#pragma unroll
      for (int c = 0; c < 4; ++c)
#pragma unroll
        for (int i = 0; i < 8; ++i) {
          int m  = rowBase + 32 * wm + 16 * r + i + 8 * lh;
          int nc = colBase + 64 * wn + 16 * c + ln;
          if (m < M && nc < Nc) {
            if (OUT_BF16)
              ((uint16_t*)Cout)[(size_t)m * Nc + nc] = f2bf(acc[r][c][i]);
            else
              ((float*)Cout)[(size_t)m * Nc + nc] = acc[r][c][i];
          }
        }
  }
}

// ---------------- flash attention (one wave, 16 query rows) ----------------
// Q,K,V,O: bf16 [B][n][D], head h occupies columns h*64..h*64+63
// All keys touched are < kEnd = q0+16 <= n, so loads are branch-free; the
// second half-tile of the diagonal chunk is fully masked and skipped entirely.
__global__ __launch_bounds__(32) void k_flash_attn(const uint16_t* __restrict__ Q,
                                                   const uint16_t* __restrict__ Kt,
                                                   const uint16_t* __restrict__ V,
                                                   uint16_t* __restrict__ O, int n) {
  __shared__ float lP[16 * 32];
  const int lane = threadIdx.x;
  const int ln = lane & 15, lh = lane >> 4;
  const int b = blockIdx.z, h = blockIdx.y, q0 = blockIdx.x * 16;
  const size_t base = ((size_t)b * n) * cD + (size_t)h * cDH;
  const uint16_t* Qp = Q + base;
  const uint16_t* Kp = Kt + base;
  const uint16_t* Vp = V + base;

  Frag qf[2];
  {
    const uint16_t* qp = Qp + (size_t)(q0 + ln) * cD + lh * 8;
    uint4 a = *(const uint4*)(qp);
    uint4 b4 = *(const uint4*)(qp + 16);
    uint4 c = *(const uint4*)(qp + 32);
    uint4 d = *(const uint4*)(qp + 48);
    qf[0].w[0] = a.x; qf[0].w[1] = a.y; qf[0].w[2] = a.z; qf[0].w[3] = a.w;
    qf[0].w[4] = b4.x; qf[0].w[5] = b4.y; qf[0].w[6] = b4.z; qf[0].w[7] = b4.w;
    qf[1].w[0] = c.x; qf[1].w[1] = c.y; qf[1].w[2] = c.z; qf[1].w[3] = c.w;
    qf[1].w[4] = d.x; qf[1].w[5] = d.y; qf[1].w[6] = d.z; qf[1].w[7] = d.w;
  }

  f32x8 accO[4];
#pragma unroll
  for (int c = 0; c < 4; ++c) accO[c] = zero8();
  float mrow[8], lrow[8];
#pragma unroll
  for (int i = 0; i < 8; ++i) { mrow[i] = -3.0e38f; lrow[i] = 0.0f; }

  const int kEnd = q0 + 16;   // exclusive key bound (<= n)
  for (int kb = 0; kb < kEnd; kb += 32) {
    const bool hasT1 = (kb + 16 < kEnd);  // wave-uniform
    const bool diag  = (kb + 32 > q0);    // wave-uniform
    f32x8 s0 = zero8(), s1 = zero8();
    {
      const uint16_t* kp = Kp + (size_t)(kb + ln) * cD + lh * 8;
      uint4 a = *(const uint4*)(kp);
      uint4 b4 = *(const uint4*)(kp + 16);
      uint4 c = *(const uint4*)(kp + 32);
      uint4 d = *(const uint4*)(kp + 48);
      Frag k0f, k1f;
      k0f.w[0] = a.x; k0f.w[1] = a.y; k0f.w[2] = a.z; k0f.w[3] = a.w;
      k0f.w[4] = b4.x; k0f.w[5] = b4.y; k0f.w[6] = b4.z; k0f.w[7] = b4.w;
      k1f.w[0] = c.x; k1f.w[1] = c.y; k1f.w[2] = c.z; k1f.w[3] = c.w;
      k1f.w[4] = d.x; k1f.w[5] = d.y; k1f.w[6] = d.z; k1f.w[7] = d.w;
      s0 = __builtin_amdgcn_wmma_f32_16x16x32_bf16(false, qf[0].v, false, k0f.v, (short)0, s0,
                                                   false, false);
      s0 = __builtin_amdgcn_wmma_f32_16x16x32_bf16(false, qf[1].v, false, k1f.v, (short)0, s0,
                                                   false, false);
    }
    if (hasT1) {
      const uint16_t* kp = Kp + (size_t)(kb + 16 + ln) * cD + lh * 8;
      uint4 a = *(const uint4*)(kp);
      uint4 b4 = *(const uint4*)(kp + 16);
      uint4 c = *(const uint4*)(kp + 32);
      uint4 d = *(const uint4*)(kp + 48);
      Frag k0f, k1f;
      k0f.w[0] = a.x; k0f.w[1] = a.y; k0f.w[2] = a.z; k0f.w[3] = a.w;
      k0f.w[4] = b4.x; k0f.w[5] = b4.y; k0f.w[6] = b4.z; k0f.w[7] = b4.w;
      k1f.w[0] = c.x; k1f.w[1] = c.y; k1f.w[2] = c.z; k1f.w[3] = c.w;
      k1f.w[4] = d.x; k1f.w[5] = d.y; k1f.w[6] = d.z; k1f.w[7] = d.w;
      s1 = __builtin_amdgcn_wmma_f32_16x16x32_bf16(false, qf[0].v, false, k0f.v, (short)0, s1,
                                                   false, false);
      s1 = __builtin_amdgcn_wmma_f32_16x16x32_bf16(false, qf[1].v, false, k1f.v, (short)0, s1,
                                                   false, false);
    }
    // scale + causal mask + online softmax (rows 0..7 in lanes 0-15, 8..15 in 16-31)
#pragma unroll
    for (int i = 0; i < 8; ++i) {
      float v0 = s0[i] * 0.125f;
      float v1 = hasT1 ? s1[i] * 0.125f : -1e9f;
      if (diag) {
        int row = q0 + i + 8 * lh;
        if (kb + ln > row) v0 = -1e9f;
        if (kb + 16 + ln > row) v1 = -1e9f;
      }
      float mnew  = fmaxf(mrow[i], redmax16(fmaxf(v0, v1)));
      float p0    = __expf(v0 - mnew);
      float p1    = __expf(v1 - mnew);
      float alpha = __expf(mrow[i] - mnew);
      lrow[i] = lrow[i] * alpha + redsum16(p0 + p1);
      mrow[i] = mnew;
#pragma unroll
      for (int c = 0; c < 4; ++c) accO[c][i] *= alpha;
      lP[(i + 8 * lh) * 32 + ln]      = p0;
      lP[(i + 8 * lh) * 32 + 16 + ln] = p1;
    }
    __syncthreads();
    // P as A-fragment (16x32, bf16)
    Frag pf;
#pragma unroll
    for (int j = 0; j < 8; ++j) {
      int kk = (j >> 2) * 16 + lh * 8 + (j & 3) * 2;
      pf.h[2 * j]     = f2bf(lP[ln * 32 + kk]);
      pf.h[2 * j + 1] = f2bf(lP[ln * 32 + kk + 1]);
    }
    // P @ V  (clamped addresses; masked keys have P == 0)
#pragma unroll
    for (int c = 0; c < 4; ++c) {
      Frag vf;
      int d = c * 16 + ln;
#pragma unroll
      for (int j = 0; j < 8; ++j) {
        int krel = (j >> 2) * 16 + lh * 8 + (j & 3) * 2;
        int key = kb + krel;
        int k0c = (key < kEnd) ? key : (kEnd - 1);
        int k1c = (key + 1 < kEnd) ? (key + 1) : (kEnd - 1);
        vf.h[2 * j]     = Vp[(size_t)k0c * cD + d];
        vf.h[2 * j + 1] = Vp[(size_t)k1c * cD + d];
      }
      accO[c] = __builtin_amdgcn_wmma_f32_16x16x32_bf16(
          false, pf.v, false, vf.v, (short)0, accO[c], false, false);
    }
    __syncthreads();
  }

#pragma unroll
  for (int i = 0; i < 8; ++i) {
    float inv = 1.0f / lrow[i];
    int row = q0 + i + 8 * lh;
#pragma unroll
    for (int c = 0; c < 4; ++c)
      O[base + (size_t)row * cD + c * 16 + ln] = f2bf(accO[c][i] * inv);
  }
}

// ---------------- Gram matrix of the 3 scale outputs ----------------
__global__ __launch_bounds__(256) void k_zero_g(float* G) {
  if (threadIdx.x < 16 && blockIdx.x == 0) G[threadIdx.x] = 0.0f;
}

__global__ __launch_bounds__(256) void k_gram(const float* __restrict__ Z0,
                                              const float* __restrict__ Z1,
                                              const float* __restrict__ Z2,
                                              float* __restrict__ G) {
  __shared__ float red[256];
  float a[6] = {0, 0, 0, 0, 0, 0};
  size_t stride = (size_t)gridDim.x * blockDim.x;
  for (size_t i = (size_t)blockIdx.x * blockDim.x + threadIdx.x; i < cBND; i += stride) {
    size_t b = i / ((size_t)cN * cD);
    size_t rem = i - b * (size_t)cN * cD;
    size_t r = rem / cD, d = rem - r * cD;
    float z0 = Z0[i];
    float z1 = Z1[(b * (cN / 2) + (r >> 1)) * cD + d];
    float z2 = Z2[(b * (cN / 4) + (r >> 2)) * cD + d];
    a[0] += z0 * z0; a[1] += z0 * z1; a[2] += z0 * z2;
    a[3] += z1 * z1; a[4] += z1 * z2; a[5] += z2 * z2;
  }
  for (int k = 0; k < 6; ++k) {
    red[threadIdx.x] = a[k];
    __syncthreads();
    for (int s = 128; s > 0; s >>= 1) {
      if ((int)threadIdx.x < s) red[threadIdx.x] += red[threadIdx.x + s];
      __syncthreads();
    }
    if (threadIdx.x == 0) atomicAdd(&G[k], red[0]);
    __syncthreads();
  }
}

// ---------------- scalar kernel: Nash weights, aux loss, Sinkhorn ----------------
DEVINL void softmax3(const float* in, float* o) {
  float m = fmaxf(in[0], fmaxf(in[1], in[2]));
  float e0 = __expf(in[0] - m), e1 = __expf(in[1] - m), e2 = __expf(in[2] - m);
  float s = e0 + e1 + e2;
  o[0] = e0 / s; o[1] = e1 / s; o[2] = e2 / s;
}

DEVINL void sinkhorn2(const float* lg, float* p00, float* p01) {
  float M[2][2];
  for (int i = 0; i < 2; ++i)
    for (int j = 0; j < 2; ++j) M[i][j] = __expf(lg[i * 2 + j]);
  for (int it = 0; it < 10; ++it) {
    for (int i = 0; i < 2; ++i) {
      float s = M[i][0] + M[i][1];
      M[i][0] /= s; M[i][1] /= s;
    }
    for (int j = 0; j < 2; ++j) {
      float s = M[0][j] + M[1][j];
      M[0][j] /= s; M[1][j] /= s;
    }
  }
  *p00 = M[0][0]; *p01 = M[0][1];
}

__global__ __launch_bounds__(32) void k_scalars(const float* __restrict__ G6,
                                                const float* __restrict__ agg,
                                                const float* __restrict__ mhcA,
                                                const float* __restrict__ mhcF,
                                                float* __restrict__ S,
                                                float* __restrict__ auxOut) {
  if (threadIdx.x != 0 || blockIdx.x != 0) return;
  const float inv = 1.0f / (float)cBND;
  float G[3][3];
  G[0][0] = G6[0] * inv;
  G[0][1] = G[1][0] = G6[1] * inv;
  G[0][2] = G[2][0] = G6[2] * inv;
  G[1][1] = G6[3] * inv;
  G[1][2] = G[2][1] = G6[4] * inv;
  G[2][2] = G6[5] * inv;
  float ag[3] = {agg[0], agg[1], agg[2]};
  float w[3];
  softmax3(ag, w);
  for (int it = 0; it < 3; ++it) {
    float q = 0.0f;
    for (int j = 0; j < 3; ++j)
      for (int k = 0; k < 3; ++k) q += w[j] * w[k] * G[j][k];
    float lg[3];
    for (int l = 0; l < 3; ++l) {
      float t = 0.0f;
      for (int j = 0; j < 3; ++j) t += w[j] * G[l][j];
      lg[l] = ag[l] - (G[l][l] - 2.0f * t + q);   // agg + u
    }
    softmax3(lg, w);
  }
  float aux = 0.0f;
  for (int l = 0; l < 3; ++l) aux += w[l] * __logf(w[l] * 3.0f + 1e-9f);
  S[0] = w[0]; S[1] = w[1]; S[2] = w[2];
  float lgA[4] = {mhcA[0], mhcA[1], mhcA[2], mhcA[3]};
  float lgF[4] = {mhcF[0], mhcF[1], mhcF[2], mhcF[3]};
  sinkhorn2(lgA, &S[3], &S[4]);
  sinkhorn2(lgF, &S[5], &S[6]);
  *auxOut = aux;
}

// ---------------- residual mixes and FFN elementwise ----------------
__global__ __launch_bounds__(256) void k_mix1(const float* __restrict__ x,
                                              const float* __restrict__ Z0,
                                              const float* __restrict__ Z1,
                                              const float* __restrict__ Z2,
                                              const float* __restrict__ S,
                                              float* __restrict__ x1) {
  float w0 = S[0], w1 = S[1], w2 = S[2], p00 = S[3], p01 = S[4];
  size_t stride = (size_t)gridDim.x * blockDim.x;
  for (size_t i = (size_t)blockIdx.x * blockDim.x + threadIdx.x; i < cBND; i += stride) {
    size_t b = i / ((size_t)cN * cD);
    size_t rem = i - b * (size_t)cN * cD;
    size_t r = rem / cD, d = rem - r * cD;
    float ybar = w0 * Z0[i] +
                 w1 * Z1[(b * (cN / 2) + (r >> 1)) * cD + d] +
                 w2 * Z2[(b * (cN / 4) + (r >> 2)) * cD + d];
    x1[i] = p00 * x[i] + p01 * ybar;
  }
}

__global__ __launch_bounds__(256) void k_silu_mul(const float* __restrict__ g,
                                                  const float* __restrict__ u,
                                                  uint16_t* __restrict__ hb) {
  size_t total = (size_t)cBN * cHIDP;
  size_t stride = (size_t)gridDim.x * blockDim.x;
  for (size_t i = (size_t)blockIdx.x * blockDim.x + threadIdx.x; i < total; i += stride) {
    size_t row = i / cHIDP;
    int j = (int)(i - row * cHIDP);
    if (j < cHID) {
      float gv = g[row * cHID + j];
      float uv = u[row * cHID + j];
      float sv = gv / (1.0f + __expf(-gv));
      hb[i] = f2bf(sv * uv);
    } else {
      hb[i] = 0;
    }
  }
}

__global__ __launch_bounds__(256) void k_mix2(const float* __restrict__ x1,
                                              const float* __restrict__ ffn,
                                              const float* __restrict__ S,
                                              float* __restrict__ out) {
  float p00 = S[5], p01 = S[6];
  size_t stride = (size_t)gridDim.x * blockDim.x;
  for (size_t i = (size_t)blockIdx.x * blockDim.x + threadIdx.x; i < cBND; i += stride)
    out[i] = p00 * x1[i] + p01 * ffn[i];
}

// ---------------- host launcher ----------------
extern "C" void kernel_launch(void* const* d_in, const int* in_sizes, int n_in,
                              void* d_out, int out_size, void* d_ws, size_t ws_size,
                              hipStream_t stream) {
  (void)in_sizes; (void)n_in; (void)out_size; (void)ws_size;
  const float* x    = (const float*)d_in[0];
  const float* nAw  = (const float*)d_in[1];
  const float* nFw  = (const float*)d_in[2];
  const float* Wdec = (const float*)d_in[3];
  const float* Wq   = (const float*)d_in[4];
  const float* Wk   = (const float*)d_in[5];
  const float* Wv   = (const float*)d_in[6];
  const float* Wo   = (const float*)d_in[7];
  const float* agg  = (const float*)d_in[8];
  const float* wg   = (const float*)d_in[9];
  const float* wu   = (const float*)d_in[10];
  const float* wd   = (const float*)d_in[11];
  const float* mhcA = (const float*)d_in[12];
  const float* mhcF = (const float*)d_in[13];
  float* out = (float*)d_out;

  char* W = (char*)d_ws;
  auto U16 = [&](size_t off) { return (uint16_t*)(W + off); };
  auto F32 = [&](size_t off) { return (float*)(W + off); };
  auto nb  = [](size_t n) { unsigned b = (unsigned)((n + 255) / 256); return b > 8192u ? 8192u : b; };

  // 1) weight conversions (f32 -> bf16)
  k_conv_bf16<<<nb((size_t)3 * cD * cD), 256, 0, stream>>>(Wdec, U16(OFF_WDEC), (size_t)3 * cD * cD);
  k_conv_bf16<<<nb((size_t)3 * cD * cD), 256, 0, stream>>>(Wq, U16(OFF_WQ), (size_t)3 * cD * cD);
  k_conv_bf16<<<nb((size_t)3 * cD * cD), 256, 0, stream>>>(Wk, U16(OFF_WK), (size_t)3 * cD * cD);
  k_conv_bf16<<<nb((size_t)cD * cD), 256, 0, stream>>>(Wv, U16(OFF_WV), (size_t)cD * cD);
  k_conv_bf16<<<nb((size_t)cD * cD), 256, 0, stream>>>(Wo, U16(OFF_WO), (size_t)cD * cD);
  k_conv_bf16<<<nb((size_t)cD * cHID), 256, 0, stream>>>(wg, U16(OFF_WG), (size_t)cD * cHID);
  k_conv_bf16<<<nb((size_t)cD * cHID), 256, 0, stream>>>(wu, U16(OFF_WU), (size_t)cD * cHID);
  k_conv_pad_rows<<<nb((size_t)cHIDP * cD), 256, 0, stream>>>(wd, U16(OFF_WD), cHID, cD, cHIDP);

  // 2) RMSNorm (attn) -> bf16
  k_rmsnorm_bf16<<<cBN, 256, 0, stream>>>(x, nAw, U16(OFF_XN0));

  // 3) pooled scales
  k_pool2<<<nb((size_t)cB * 1024 * cD), 256, 0, stream>>>(U16(OFF_XN0), U16(OFF_S1), 1024,
                                                          (size_t)cB * 1024 * cD);
  k_pool2<<<nb((size_t)cB * 512 * cD), 256, 0, stream>>>(U16(OFF_S1), U16(OFF_S2), 512,
                                                         (size_t)cB * 512 * cD);

  const int nl[3]     = {2048, 1024, 512};
  const size_t roff[3] = {0, 4096, 6144};   // token-row offsets within TOKS
  const uint16_t* scal[3] = {U16(OFF_XN0), U16(OFF_S1), U16(OFF_S2)};
  auto gg = [](int M, int Nc) { return dim3((unsigned)((M + 127) / 128), (unsigned)((Nc + 127) / 128)); };

  // 4) per-scale projections + attention + output projection
  for (int l = 0; l < 3; ++l) {
    int M = cB * nl[l];
    uint16_t* xd = U16(OFF_XDEC) + roff[l] * cD;
    uint16_t* Qp = U16(OFF_Q) + roff[l] * cD;
    uint16_t* Kp = U16(OFF_K) + roff[l] * cD;
    uint16_t* Vp = U16(OFF_V) + roff[l] * cD;
    uint16_t* Op = U16(OFF_O) + roff[l] * cD;
    float*    Zp = F32(OFF_Z) + roff[l] * cD;

    k_gemm_wmma<true><<<gg(M, cD), 256, 0, stream>>>(scal[l], U16(OFF_WDEC) + (size_t)l * cD * cD,
                                                     xd, M, cD, cD);
    k_gemm_wmma<true><<<gg(M, cD), 256, 0, stream>>>(xd, U16(OFF_WQ) + (size_t)l * cD * cD,
                                                     Qp, M, cD, cD);
    k_gemm_wmma<true><<<gg(M, cD), 256, 0, stream>>>(xd, U16(OFF_WK) + (size_t)l * cD * cD,
                                                     Kp, M, cD, cD);
    k_gemm_wmma<true><<<gg(M, cD), 256, 0, stream>>>(xd, U16(OFF_WV), Vp, M, cD, cD);

    dim3 agrid((unsigned)(nl[l] / 16), (unsigned)cH, (unsigned)cB);
    k_flash_attn<<<agrid, 32, 0, stream>>>(Qp, Kp, Vp, Op, nl[l]);

    k_gemm_wmma<false><<<gg(M, cD), 256, 0, stream>>>(Op, U16(OFF_WO), (void*)Zp, M, cD, cD);
  }

  // 5) Nash aggregation via 3x3 Gram + tiny scalar kernel (also Sinkhorns + aux loss)
  k_zero_g<<<1, 32, 0, stream>>>(F32(OFF_G));
  k_gram<<<1024, 256, 0, stream>>>(F32(OFF_Z), F32(OFF_Z) + (size_t)4096 * cD,
                                   F32(OFF_Z) + (size_t)6144 * cD, F32(OFF_G));
  k_scalars<<<1, 32, 0, stream>>>(F32(OFF_G), agg, mhcA, mhcF, F32(OFF_SC), out + cBND);

  // 6) mHC residual mix (attention branch)
  k_mix1<<<nb(cBND), 256, 0, stream>>>(x, F32(OFF_Z), F32(OFF_Z) + (size_t)4096 * cD,
                                       F32(OFF_Z) + (size_t)6144 * cD, F32(OFF_SC), F32(OFF_X1));

  // 7) FFN: RMSNorm -> gate/up GEMMs -> silu*up -> down GEMM
  k_rmsnorm_bf16<<<cBN, 256, 0, stream>>>(F32(OFF_X1), nFw, U16(OFF_XN2));
  k_gemm_wmma<false><<<gg(cBN, cHID), 256, 0, stream>>>(U16(OFF_XN2), U16(OFF_WG),
                                                        (void*)F32(OFF_GATE), cBN, cD, cHID);
  k_gemm_wmma<false><<<gg(cBN, cHID), 256, 0, stream>>>(U16(OFF_XN2), U16(OFF_WU),
                                                        (void*)F32(OFF_UP), cBN, cD, cHID);
  k_silu_mul<<<nb((size_t)cBN * cHIDP), 256, 0, stream>>>(F32(OFF_GATE), F32(OFF_UP), U16(OFF_HB));
  k_gemm_wmma<false><<<gg(cBN, cD), 256, 0, stream>>>(U16(OFF_HB), U16(OFF_WD),
                                                      (void*)F32(OFF_FFN), cBN, cHIDP, cD);

  // 8) final mHC mix -> output
  k_mix2<<<nb(cBND), 256, 0, stream>>>(F32(OFF_X1), F32(OFF_FFN), F32(OFF_SC), out);
}
